// GraphTransformerDSTA_21071109554593
// MI455X (gfx1250) — matrix operand order repaired and checked by reading, not compiled, not run on hardware
//
#include <hip/hip_runtime.h>
#include <hip/hip_bf16.h>
#include <math.h>

typedef __attribute__((ext_vector_type(16))) _Float16 v16h;
typedef __attribute__((ext_vector_type(8)))  float    v8f;

#define NNODES 2000
#define NEDGES 32000
#define BB 8
#define TT 12
#define BT 96            // B*T
#define HD 64
#define HS 768           // T*H
#define NHEADS 8
#define MBT 192000       // B*T*NNODES
#define MB  16000        // B*NNODES

// ---------------------------------------------------------------- utilities
__global__ void zero_f32(float* __restrict__ p, long n) {
  long i = (long)blockIdx.x * blockDim.x + threadIdx.x;
  long stride = (long)gridDim.x * blockDim.x;
  for (; i < n; i += stride) p[i] = 0.f;
}

// Pack a row-major K x Nact f32 weight matrix into WMMA B-fragment order:
// tiles (kt,nt); within a tile, 32 lanes x 16 halves.
// B lane layout (dense 16-bit B 32x16): lanes 0-15 hold K=0..7,16..23,
// lanes 16-31 hold K=8..15,24..31; column n == lane & 15.
__global__ void pack_b_f16(const float* __restrict__ W, _Float16* __restrict__ P,
                           int K, int Nact, int NT) {
  int idx = blockIdx.x * blockDim.x + threadIdx.x;
  int total = (K >> 5) * NT * 512;
  if (idx >= total) return;
  int j    = idx & 15;
  int lane = (idx >> 4) & 31;
  int nt   = (idx >> 9) % NT;
  int kt   = idx / (NT * 512);
  int n = nt * 16 + (lane & 15);
  int k = kt * 32 + ((j >> 3) << 4) + ((lane >> 4) << 3) + (j & 7);
  float v = (n < Nact) ? W[(long)k * Nact + n] : 0.f;
  P[idx] = (_Float16)v;
}

// ---------------------------------------------------------------- embedding + RoPE
__global__ void embed_rope(const float* __restrict__ Xin, const float* __restrict__ We,
                           const float* __restrict__ be, float* __restrict__ X) {
  int idx = blockIdx.x * blockDim.x + threadIdx.x;     // (bt, n, i<32)
  if (idx >= BT * NNODES * 32) return;
  int i  = idx & 31;
  int n  = (idx >> 5) % NNODES;
  int bt = idx / (NNODES * 32);
  int t  = bt % TT;
  long r2 = (long)bt * NNODES + n;
  float x0 = Xin[r2 * 2 + 0];
  float x1 = Xin[r2 * 2 + 1];
  int jj = i + 32;
  float e1 = x0 * We[i]  + x1 * We[64 + i]  + be[i];
  float e2 = x0 * We[jj] + x1 * We[64 + jj] + be[jj];
  float invf = powf(10000.f, -((float)i) * (1.f / 32.f));
  float ang = (float)t * invf;
  float c = cosf(ang), s = sinf(ang);
  float* o = X + r2 * 64;
  o[i]  = e1 * c - e2 * s;
  o[jj] = e1 * s + e2 * c;
}

// ---------------------------------------------------------------- WMMA GEMM
// C(M x Nstore) = A(M x K, f32 row-major) * Bp(packed f16) [+ bias] [relu]
// One wave computes a 16 x (16*NTW) strip: the A fragment is loaded once per
// K-step and fanned across NTW adjacent B tiles (contiguous in packed layout),
// giving NTW WMMAs per 4+2*NTW b128 loads and cutting redundant A traffic by NTW.
template <int NTW>
__global__ __launch_bounds__(256) void gemm_wmma_f16(
    const float* __restrict__ A, const _Float16* __restrict__ Bp,
    const float* __restrict__ bias, float* __restrict__ C,
    int M, int K, int Npad, int Nstore, int relu) {
  int wid  = (int)((blockIdx.x * blockDim.x + threadIdx.x) >> 5);
  int lane = threadIdx.x & 31;
  int tilesN = Npad >> 4;
  int groups = tilesN / NTW;
  int tiles  = (M >> 4) * groups;
  if (wid >= tiles) return;                 // wave-uniform: EXEC all-ones inside
  int tm = wid / groups, tg = wid - tm * groups;
  int row   = tm * 16 + (lane & 15);
  int khalf = lane >> 4;                    // which K-half this lane covers
  const float* arow = A + (long)row * K + khalf * 8;
  const _Float16* bptr = Bp + ((long)(tg * NTW) * 32 + lane) * 16;
  long bstride = (long)tilesN * 512;        // halves between kt tiles
  v8f acc[NTW];
  #pragma unroll
  for (int j = 0; j < NTW; ++j) acc[j] = (v8f){};
  for (int k0 = 0; k0 < K; k0 += 32) {
    // A fragment: lane holds K in [k0+khalf*8, +8) and [k0+16+khalf*8, +8)
    float4 a0 = *(const float4*)(arow + k0 + 0);
    float4 a1 = *(const float4*)(arow + k0 + 4);
    float4 a2 = *(const float4*)(arow + k0 + 16);
    float4 a3 = *(const float4*)(arow + k0 + 20);
    v16h af;
    af[0]=(_Float16)a0.x;  af[1]=(_Float16)a0.y;  af[2]=(_Float16)a0.z;  af[3]=(_Float16)a0.w;
    af[4]=(_Float16)a1.x;  af[5]=(_Float16)a1.y;  af[6]=(_Float16)a1.z;  af[7]=(_Float16)a1.w;
    af[8]=(_Float16)a2.x;  af[9]=(_Float16)a2.y;  af[10]=(_Float16)a2.z; af[11]=(_Float16)a2.w;
    af[12]=(_Float16)a3.x; af[13]=(_Float16)a3.y; af[14]=(_Float16)a3.z; af[15]=(_Float16)a3.w;
    #pragma unroll
    for (int j = 0; j < NTW; ++j) {
      v16h bf = *(const v16h*)(bptr + j * 512);   // adjacent nt tiles contiguous
      acc[j] = __builtin_amdgcn_wmma_f32_16x16x32_f16(
          false, af, false, bf, (short)0, acc[j], false, false);
    }
    bptr += bstride;
  }
  int mbase = tm * 16 + (khalf << 3);       // lanes>=16 hold rows M=8..15
  #pragma unroll
  for (int j = 0; j < NTW; ++j) {
    int nout = (tg * NTW + j) * 16 + (lane & 15);
    if (nout < Nstore) {                    // only false for padded decoder tile
      float bb = bias ? bias[nout] : 0.f;
      #pragma unroll
      for (int r = 0; r < 8; ++r) {
        float v = acc[j][r] + bb;
        if (relu) v = fmaxf(v, 0.f);
        C[(long)(mbase + r) * Nstore + nout] = v;
      }
    }
  }
}

// ---------------------------------------------------------------- edge attention
__global__ void edge_logits(const float* __restrict__ q, const float* __restrict__ k,
                            const int* __restrict__ src, const int* __restrict__ dst,
                            float* __restrict__ lg) {
  int idx = blockIdx.x * blockDim.x + threadIdx.x;     // (e, bt, h)
  if (idx >= NEDGES * BT * NHEADS) return;
  int h  = idx & 7;
  int bt = (idx >> 3) % BT;
  int e  = idx / (BT * NHEADS);
  int s = src[e], d = dst[e];
  const float* qp = q + ((long)bt * NNODES + d) * 64 + h * 8;
  const float* kp = k + ((long)bt * NNODES + s) * 64 + h * 8;
  float acc = 0.f;
  #pragma unroll
  for (int i = 0; i < 8; ++i) acc += qp[i] * kp[i];
  lg[idx] = acc * 0.35355339059327373f;                // 1/sqrt(8)
}

__global__ void edge_expsum(float* __restrict__ lg, const int* __restrict__ dst,
                            float* __restrict__ den) {
  int idx = blockIdx.x * blockDim.x + threadIdx.x;
  if (idx >= NEDGES * BT * NHEADS) return;
  int h  = idx & 7;
  int bt = (idx >> 3) % BT;
  int e  = idx / (BT * NHEADS);
  float ex = expf(lg[idx]);   // logits are tiny; max-subtraction cancels exactly
  lg[idx] = ex;
  atomicAdd(&den[((long)dst[e] * BT + bt) * NHEADS + h], ex);
}

__global__ void edge_aggregate(const float* __restrict__ lg, const float* __restrict__ den,
                               const float* __restrict__ v, const int* __restrict__ src,
                               const int* __restrict__ dst, const float* __restrict__ ew,
                               float* __restrict__ agg) {
  int idx = blockIdx.x * blockDim.x + threadIdx.x;
  if (idx >= NEDGES * BT * NHEADS) return;
  int h  = idx & 7;
  int bt = (idx >> 3) % BT;
  int e  = idx / (BT * NHEADS);
  int s = src[e], d = dst[e];
  float alpha = lg[idx] / den[((long)d * BT + bt) * NHEADS + h];
  float w = alpha * ew[e];
  const float* vp = v + ((long)bt * NNODES + s) * 64 + h * 8;
  float* ap = agg + ((long)bt * NNODES + d) * 64 + h * 8;
  #pragma unroll
  for (int i = 0; i < 8; ++i) atomicAdd(&ap[i], vp[i] * w);
}

// ---------------------------------------------------------------- transposes + LayerNorm
// (B,T,N,64) -> rows (b*N+n, 768), optional LayerNorm. One block per row.
__global__ void t_bn_ln(const float* __restrict__ Y, const float* __restrict__ g,
                        const float* __restrict__ b, float* __restrict__ O, int do_ln) {
  int rowid = blockIdx.x;                    // b*NNODES + n
  int bb = rowid / NNODES, n = rowid - bb * NNODES;
  int tid = threadIdx.x;                     // 256 threads, 3 elems each
  float vals[3];
  #pragma unroll
  for (int i = 0; i < 3; ++i) {
    int c = tid + i * 256;
    int t = c >> 6, f = c & 63;
    vals[i] = Y[(((long)bb * TT + t) * NNODES + n) * 64 + f];
  }
  float* orow = O + (long)rowid * HS;
  if (!do_ln) {
    #pragma unroll
    for (int i = 0; i < 3; ++i) orow[tid + i * 256] = vals[i];
    return;
  }
  __shared__ float s1[256], s2[256];
  s1[tid] = vals[0] + vals[1] + vals[2];
  s2[tid] = vals[0]*vals[0] + vals[1]*vals[1] + vals[2]*vals[2];
  __syncthreads();
  for (int st = 128; st > 0; st >>= 1) {
    if (tid < st) { s1[tid] += s1[tid + st]; s2[tid] += s2[tid + st]; }
    __syncthreads();
  }
  float mean = s1[0] * (1.f / 768.f);
  float var  = s2[0] * (1.f / 768.f) - mean * mean;
  float rs = rsqrtf(var + 1e-5f);
  #pragma unroll
  for (int i = 0; i < 3; ++i) {
    int c = tid + i * 256;
    orow[c] = (vals[i] - mean) * rs * g[c] + b[c];
  }
}

// rows (b*N+n, 768) -> LayerNorm -> scatter back to (B,T,N,64)
__global__ void ln_scatter(const float* __restrict__ Y2, const float* __restrict__ g,
                           const float* __restrict__ b, float* __restrict__ X) {
  int rowid = blockIdx.x;
  int bb = rowid / NNODES, n = rowid - bb * NNODES;
  int tid = threadIdx.x;
  const float* yrow = Y2 + (long)rowid * HS;
  float vals[3];
  #pragma unroll
  for (int i = 0; i < 3; ++i) vals[i] = yrow[tid + i * 256];
  __shared__ float s1[256], s2[256];
  s1[tid] = vals[0] + vals[1] + vals[2];
  s2[tid] = vals[0]*vals[0] + vals[1]*vals[1] + vals[2]*vals[2];
  __syncthreads();
  for (int st = 128; st > 0; st >>= 1) {
    if (tid < st) { s1[tid] += s1[tid + st]; s2[tid] += s2[tid + st]; }
    __syncthreads();
  }
  float mean = s1[0] * (1.f / 768.f);
  float var  = s2[0] * (1.f / 768.f) - mean * mean;
  float rs = rsqrtf(var + 1e-5f);
  #pragma unroll
  for (int i = 0; i < 3; ++i) {
    int c = tid + i * 256;
    int t = c >> 6, f = c & 63;
    X[(((long)bb * TT + t) * NNODES + n) * 64 + f] = (vals[i] - mean) * rs * g[c] + b[c];
  }
}

// decoder rows (b*N+n, 24) -> out (B, HORIZON, N, F_IN)
__global__ void reorder_out(const float* __restrict__ dec, float* __restrict__ out) {
  int idx = blockIdx.x * blockDim.x + threadIdx.x;
  if (idx >= BB * TT * NNODES * 2) return;
  int f = idx & 1;
  int n = (idx >> 1) % NNODES;
  int t = (idx / (NNODES * 2)) % TT;
  int b = idx / (TT * NNODES * 2);
  out[idx] = dec[((long)b * NNODES + n) * 24 + t * 2 + f];
}

// ---------------------------------------------------------------- launcher
extern "C" void kernel_launch(void* const* d_in, const int* in_sizes, int n_in,
                              void* d_out, int out_size, void* d_ws, size_t ws_size,
                              hipStream_t stream) {
  const float* x_in = (const float*)d_in[0];
  const int*   ei   = (const int*)  d_in[1];
  const float* ew   = (const float*)d_in[2];
  const float* We   = (const float*)d_in[3];
  const float* be   = (const float*)d_in[4];
  const float* Wq   = (const float*)d_in[5];
  const float* Wk   = (const float*)d_in[6];
  const float* Wv   = (const float*)d_in[7];
  const float* Wo   = (const float*)d_in[8];
  const float* ln1g = (const float*)d_in[9];
  const float* ln1b = (const float*)d_in[10];
  const float* W1   = (const float*)d_in[11];
  const float* b1   = (const float*)d_in[12];
  const float* W2   = (const float*)d_in[13];
  const float* b2   = (const float*)d_in[14];
  const float* ln2g = (const float*)d_in[15];
  const float* ln2b = (const float*)d_in[16];
  const float* Wd   = (const float*)d_in[17];
  const float* bd   = (const float*)d_in[18];

  const int* src  = ei;
  const int* dstn = ei + NEDGES;

  // workspace arena (bytes, all 256-aligned)
  char* ws = (char*)d_ws;
  float* S0  = (float*)(ws + 0L);            // 12,288,000 f32 each
  float* S1  = (float*)(ws + 49152000L);
  float* S2  = (float*)(ws + 98304000L);
  float* S3  = (float*)(ws + 147456000L);
  float* EXL = (float*)(ws + 196608000L);    // 24,576,000 f32 (edge exp-logits)
  float* DEN = (float*)(ws + 294912000L);    // 1,536,000 f32
  float* DEC = (float*)(ws + 301056000L);    // 384,000 f32 (stride 24)
  _Float16* PK = (_Float16*)(ws + 302592000L);

  _Float16* pQ[2]  = {PK + 0,     PK + 4096};
  _Float16* pKk[2] = {PK + 8192,  PK + 12288};
  _Float16* pV[2]  = {PK + 16384, PK + 20480};
  _Float16* pO[2]  = {PK + 24576, PK + 28672};
  _Float16* pW1[2] = {PK + 32768,              PK + 32768 + 589824};
  _Float16* pW2[2] = {PK + 32768 + 2*589824,   PK + 32768 + 3*589824};
  _Float16* pD     =  PK + 32768 + 4*589824;

  auto gemm = [&](const float* A, const _Float16* Bp, const float* bias, float* C,
                  int M, int K, int Npad, int Nstore, int relu) {
    int tilesN = Npad >> 4;
    if ((tilesN & 3) == 0) {
      int tiles = (M >> 4) * (tilesN >> 2);
      gemm_wmma_f16<4><<<(tiles + 7) / 8, 256, 0, stream>>>(A, Bp, bias, C, M, K, Npad, Nstore, relu);
    } else {
      int tiles = (M >> 4) * (tilesN >> 1);
      gemm_wmma_f16<2><<<(tiles + 7) / 8, 256, 0, stream>>>(A, Bp, bias, C, M, K, Npad, Nstore, relu);
    }
  };

  // ---- pack all weights to f16 WMMA-B layout (tiny, once per call)
  for (int l = 0; l < 2; ++l) {
    pack_b_f16<<<16,   256, 0, stream>>>(Wq + l*4096,   pQ[l],  64, 64, 4);
    pack_b_f16<<<16,   256, 0, stream>>>(Wk + l*4096,   pKk[l], 64, 64, 4);
    pack_b_f16<<<16,   256, 0, stream>>>(Wv + l*4096,   pV[l],  64, 64, 4);
    pack_b_f16<<<16,   256, 0, stream>>>(Wo + l*4096,   pO[l],  64, 64, 4);
    pack_b_f16<<<2304, 256, 0, stream>>>(W1 + (long)l*589824, pW1[l], HS, HS, 48);
    pack_b_f16<<<2304, 256, 0, stream>>>(W2 + (long)l*589824, pW2[l], HS, HS, 48);
  }
  pack_b_f16<<<96, 256, 0, stream>>>(Wd, pD, HS, 24, 2);   // N padded 24 -> 32

  // ---- embedding + RoPE -> S0 (B,T,N,64)
  embed_rope<<<24000, 256, 0, stream>>>(x_in, We, be, S0);

  const int EG = (NEDGES * BT * NHEADS + 255) / 256;       // 96000 blocks

  for (int l = 0; l < 2; ++l) {
    // QKV projections (192000x64 @ 64x64)
    gemm(S0, pQ[l],  nullptr, S1, MBT, 64, 64, 64, 0);     // Q -> S1
    gemm(S0, pKk[l], nullptr, S2, MBT, 64, 64, 64, 0);     // K -> S2
    gemm(S0, pV[l],  nullptr, S3, MBT, 64, 64, 64, 0);     // V -> S3

    // edge attention (segment softmax over dst)
    edge_logits<<<EG, 256, 0, stream>>>(S1, S2, src, dstn, EXL);
    zero_f32<<<1024, 256, 0, stream>>>(DEN, 1536000L);
    edge_expsum<<<EG, 256, 0, stream>>>(EXL, dstn, DEN);
    zero_f32<<<4096, 256, 0, stream>>>(S2, 12288000L);     // agg reuses K's slot
    edge_aggregate<<<EG, 256, 0, stream>>>(EXL, DEN, S3, src, dstn, ew, S2);

    // output projection, transpose + LN1
    gemm(S2, pO[l], nullptr, S1, MBT, 64, 64, 64, 0);      // Y -> S1
    t_bn_ln<<<MB, 256, 0, stream>>>(S1, ln1g + l*HS, ln1b + l*HS, S2, 1); // YBN -> S2

    // FFN (16000x768 @ 768x768, relu, then @ 768x768) + LN2 + scatter back
    gemm(S2, pW1[l], b1 + l*HS, S3, MB, HS, HS, HS, 1);    // H1 -> S3
    gemm(S3, pW2[l], b2 + l*HS, S1, MB, HS, HS, HS, 0);    // Y2 -> S1
    ln_scatter<<<MB, 256, 0, stream>>>(S1, ln2g + l*HS, ln2b + l*HS, S0); // -> X (S0)
  }

  // ---- decoder
  t_bn_ln<<<MB, 256, 0, stream>>>(S0, nullptr, nullptr, S1, 0);  // Z -> S1 (no LN)
  gemm(S1, pD, bd, DEC, MB, HS, 32, 24, 0);                      // Z @ Wd + bd
  reorder_out<<<1500, 256, 0, stream>>>(DEC, (float*)d_out);
}
